// RNNwavefunction_59244778881182
// MI455X (gfx1250) — matrix operand order repaired and checked by reading, not compiled, not run on hardware
//
#include <hip/hip_runtime.h>
#include <hip/hip_bf16.h>
#include <math.h>

// Problem constants (from reference)
#define BATCH 1024
#define SORB  32
#define HDIM  256
#define H3    768            // 3*H (r,z,n gates)
#define HPAD  264            // padded bf16 row stride for LDS h (528B, 16B aligned)
#define BROW  40             // staged B row stride in bf16 elements (64B data + 16B TDM pad)
#define NTHREADS 512         // 16 wave32 -> one WGP, persistent
#define PI_F 3.14159265358979323846f

typedef __attribute__((ext_vector_type(16))) __bf16 v16bf;
typedef __attribute__((ext_vector_type(8)))  __bf16 v8bf;
typedef __attribute__((ext_vector_type(8)))  float  v8f;
typedef __attribute__((ext_vector_type(4)))  unsigned int v4u;
typedef __attribute__((ext_vector_type(8)))  int  v8i;
typedef __attribute__((ext_vector_type(4)))  int  v4i;

#if __has_builtin(__builtin_amdgcn_rcpf)
#define FRCP(x) __builtin_amdgcn_rcpf(x)
#else
#define FRCP(x) (1.0f / (x))
#endif

__device__ __forceinline__ float sigmoid_f(float v) {
  return FRCP(1.0f + __expf(-v));           // v_exp_f32 + v_rcp_f32 (TRANS pipe)
}
__device__ __forceinline__ float tanh_f(float v) {
#if __has_builtin(__builtin_amdgcn_tanhf)
  return __builtin_amdgcn_tanhf(v);          // gfx1250 v_tanh_f32
#else
  float c = fminf(fmaxf(v, -10.0f), 10.0f);
  float e = __expf(2.0f * c);
  return (e - 1.0f) * FRCP(e + 1.0f);
#endif
}

// ---------------------------------------------------------------------------
// TDM: DMA one 768x32 bf16 K-chunk of W_hh (row-major 768x256) into LDS.
// D# pad: +4 dwords after every 16 dwords stored -> 80B LDS row stride.
// Issued by one wave; completion tracked with TENSORcnt.
// ---------------------------------------------------------------------------
__device__ __forceinline__ void tdm_load_whh_chunk(const __bf16* whh, unsigned int lds_off, int kc) {
  unsigned long long ga = (unsigned long long)(size_t)whh + (unsigned long long)(kc * 32 * 2);
  v4u g0;
  g0[0] = 1u;                                              // count=1, user descriptor
  g0[1] = lds_off;                                         // lds_addr (bytes)
  g0[2] = (unsigned int)(ga & 0xFFFFFFFFu);                // global_addr[31:0]
  g0[3] = ((unsigned int)(ga >> 32) & 0x01FFFFFFu) | 0x80000000u; // addr[56:32] | type=2
  v8i g1;
  g1[0] = (int)((1u << 16) | (1u << 20) | (3u << 22) | (3u << 25)); // data_size=2B, pad_en, every 16dw add 4dw
  g1[1] = (int)((HDIM & 0xFFFFu) << 16);                   // tensor_dim0=256 (lo)
  g1[2] = (int)(((HDIM >> 16) & 0xFFFFu) | ((H3 & 0xFFFFu) << 16)); // td0 hi | tensor_dim1=768 lo
  g1[3] = (int)(((H3 >> 16) & 0xFFFFu) | (32u << 16));     // td1 hi | tile_dim0=32
  g1[4] = (int)(H3 & 0xFFFFu);                             // tile_dim1=768, tile_dim2=0
  g1[5] = (int)HDIM;                                       // tensor_dim0_stride=256 (lo)
  g1[6] = 0;
  g1[7] = 0;
  v4i gz = {0, 0, 0, 0};
#if defined(__clang_major__) && (__clang_major__ >= 23)
  v8i gz8 = {0, 0, 0, 0, 0, 0, 0, 0};
  __builtin_amdgcn_tensor_load_to_lds(g0, g1, gz, gz, gz8, 0);
#else
  __builtin_amdgcn_tensor_load_to_lds(g0, g1, gz, gz, 0);
#endif
}

// ---------------------------------------------------------------------------
// Prep: convert W_hh (768x256 fp32) to bf16 in workspace (L2-resident source
// tensor for the per-step TDM stream).
// ---------------------------------------------------------------------------
__global__ void cvt_whh_bf16(const float* __restrict__ w, __bf16* __restrict__ o, int n) {
  int i = blockIdx.x * blockDim.x + threadIdx.x;
  if (i < n) o[i] = (__bf16)w[i];
}

// ---------------------------------------------------------------------------
// Persistent GRU-wavefunction kernel: 1 block, 512 threads (16 waves).
// Wave w owns h-columns [w*16, w*16+16) for all 32 rows x 3 gates:
//   6 f32 accumulators, 48 v_wmma_f32_16x16x32_bf16 per step.
// B operand double-buffered into LDS by the Tensor Data Mover.
// ---------------------------------------------------------------------------
__global__ __launch_bounds__(NTHREADS, 1)
void gru_wavefn(const int*   __restrict__ x,      // (1024,32) spins +-1
                const float* __restrict__ W_ih,   // (768,2)
                const float* __restrict__ b_ih,   // (768)
                const float* __restrict__ b_hh,   // (768)
                const float* __restrict__ fc_W,   // (2,256)
                const float* __restrict__ fc_b,   // (2)
                const __bf16* __restrict__ whh_bf,// (768,256) bf16 row-major
                float* __restrict__ out)          // (1024) complex64 as float2
{
  __shared__ __bf16 stage[2][H3 * BROW];  // 120 KB  TDM double buffer (padded rows)
  __shared__ __bf16 h_bf[SORB][HPAD];     // 16.9 KB A-operand source
  __shared__ float  h_f32[SORB][HDIM];    // 32 KB   fc head source
  __shared__ float  A0[H3], A1[H3], Bh[H3]; // 9 KB  W_ih[:,b]+b_ih selects, b_hh
  __shared__ float  fcw[2][HDIM];
  __shared__ float  fcb[2];
  __shared__ int    bits[SORB];
  __shared__ float  ysp[SORB][2][4];
  __shared__ float  red[SORB][2];

  const int tid  = threadIdx.x;
  const int wid  = tid >> 5;        // 0..15
  const int lane = tid & 31;        // wave32
  const int half = lane >> 4;
  const int lcol = lane & 15;
  const int col  = wid * 16 + lcol; // owned h-column, 0..255

  const unsigned int stage_off[2] = {
    (unsigned int)(size_t)(&stage[0][0]),   // generic addr low 32 bits == LDS byte offset
    (unsigned int)(size_t)(&stage[1][0])
  };

  // ---- one-time setup ----
  for (int c = tid; c < H3; c += NTHREADS) {
    A0[c] = W_ih[2*c + 0] + b_ih[c];
    A1[c] = W_ih[2*c + 1] + b_ih[c];
    Bh[c] = b_hh[c];
  }
  if (tid < 2 * HDIM) ((float*)fcw)[tid] = fc_W[tid];
  if (tid < 2)        fcb[tid] = fc_b[tid];
  for (int i = tid; i < SORB * HPAD; i += NTHREADS) ((__bf16*)h_bf)[i] = (__bf16)0.0f;
  __syncthreads();

  // h carried in registers: hp[m][k] = h[row = m*16 + half*8 + k][col]
  float hp[2][8];
  #pragma unroll
  for (int m = 0; m < 2; ++m)
    #pragma unroll
    for (int k = 0; k < 8; ++k) hp[m][k] = 0.0f;

  for (int t = 0; t < BATCH; ++t) {
    if (tid < SORB) bits[tid] = (1 - x[t * SORB + tid]) >> 1;
    __syncthreads();  // S1: bits ready; prev h_bf visible; prev-step stage reads done

    if (wid == 0) tdm_load_whh_chunk(whh_bf, stage_off[0], 0);   // prologue: chunk 0

    v8f ar[2], az[2], an[2];
    #pragma unroll
    for (int m = 0; m < 2; ++m) { ar[m] = (v8f)0.0f; az[m] = (v8f)0.0f; an[m] = (v8f)0.0f; }

    // ---- gh = h @ W_hh^T via WMMA, K=256 in 8 TDM-staged chunks of 32 ----
    #pragma unroll 2
    for (int kc = 0; kc < 8; ++kc) {
      if (wid == 0) __builtin_amdgcn_s_wait_tensorcnt(0);  // chunk kc landed in LDS
      __syncthreads();  // data ready for all waves; prior reads of alt buffer done
      if (wid == 0 && kc < 7)
        tdm_load_whh_chunk(whh_bf, stage_off[(kc + 1) & 1], kc + 1);  // prefetch next

      // A tiles (16x32 bf16): lanes 0-15 row=lcol K {kb..kb+7, kb+16..kb+23};
      // lanes 16-31 same rows, K shifted by 8 (ISA 16-bit A layout).
      const int akb = kc * 32 + half * 8;
      v16bf Am[2];
      #pragma unroll
      for (int m = 0; m < 2; ++m) {
        const __bf16* hr = &h_bf[m * 16 + lcol][akb];
        v8bf lo = *(const v8bf*)hr;
        v8bf hi = *(const v8bf*)(hr + 16);
        #pragma unroll
        for (int i = 0; i < 8; ++i) { Am[m][i] = lo[i]; Am[m][i + 8] = hi[i]; }
      }
      // B tiles (32x16 bf16): lane holds W_hh row n = g*256+col; K {half*8..+7, 16+half*8..+7}
      // matching A, read from the padded TDM staging rows (80B stride).
      #pragma unroll
      for (int g = 0; g < 3; ++g) {
        const __bf16* bp = &stage[kc & 1][(g * HDIM + col) * BROW + half * 8];
        v8bf lo = *(const v8bf*)bp;
        v8bf hi = *(const v8bf*)(bp + 16);
        v16bf Bv;
        #pragma unroll
        for (int i = 0; i < 8; ++i) { Bv[i] = lo[i]; Bv[i + 8] = hi[i]; }
        if (g == 0) {
          ar[0] = __builtin_amdgcn_wmma_f32_16x16x32_bf16(false, Am[0], false, Bv, (short)0, ar[0], false, false);
          ar[1] = __builtin_amdgcn_wmma_f32_16x16x32_bf16(false, Am[1], false, Bv, (short)0, ar[1], false, false);
        } else if (g == 1) {
          az[0] = __builtin_amdgcn_wmma_f32_16x16x32_bf16(false, Am[0], false, Bv, (short)0, az[0], false, false);
          az[1] = __builtin_amdgcn_wmma_f32_16x16x32_bf16(false, Am[1], false, Bv, (short)0, az[1], false, false);
        } else {
          an[0] = __builtin_amdgcn_wmma_f32_16x16x32_bf16(false, Am[0], false, Bv, (short)0, an[0], false, false);
          an[1] = __builtin_amdgcn_wmma_f32_16x16x32_bf16(false, Am[1], false, Bv, (short)0, an[1], false, false);
        }
      }
    }
    __syncthreads();  // S2: all A reads of h_bf done before overwrite

    // ---- gates; D layout: VGPR k -> row = m*16 + half*8 + k, col fixed ----
    #pragma unroll
    for (int m = 0; m < 2; ++m) {
      #pragma unroll
      for (int k = 0; k < 8; ++k) {
        const int row = m * 16 + half * 8 + k;
        const int b   = bits[row];
        const float ir  = b ? A1[col]          : A0[col];
        const float iz  = b ? A1[col +   HDIM] : A0[col +   HDIM];
        const float in_ = b ? A1[col + 2*HDIM] : A0[col + 2*HDIM];
        const float r = sigmoid_f(ir + ar[m][k] + Bh[col]);
        const float z = sigmoid_f(iz + az[m][k] + Bh[col + HDIM]);
        const float n = tanh_f(in_ + r * (an[m][k] + Bh[col + 2*HDIM]));
        const float hnew = (1.0f - z) * n + z * hp[m][k];
        hp[m][k] = hnew;
        h_bf[row][col]  = (__bf16)hnew;
        h_f32[row][col] = hnew;
      }
    }
    __syncthreads();  // S3: h_new visible

    // ---- fc head: y[s][l] = fc_b[l] + sum_c h[s][c]*fc_W[l][c], 4-way chunked ----
    if (tid < 256) {
      const int s = tid >> 3, l = (tid >> 2) & 1, ch = tid & 3;
      const int c0 = ch * 64;
      float acc = 0.0f;
      #pragma unroll 8
      for (int c = 0; c < 64; ++c) acc += h_f32[s][c0 + c] * fcw[l][c0 + c];
      ysp[s][l][ch] = acc;
    }
    __syncthreads();  // S4

    // ---- log-amplitude + phase per orbital ----
    if (tid < SORB) {
      const int s = tid;
      const float y0 = fcb[0] + ysp[s][0][0] + ysp[s][0][1] + ysp[s][0][2] + ysp[s][0][3];
      const float y1 = fcb[1] + ysp[s][1][0] + ysp[s][1][1] + ysp[s][1][2] + ysp[s][1][3];
      const float mx  = fmaxf(y0, y1);
      const float lse = mx + __logf(__expf(y0 - mx) + __expf(y1 - mx));
      const float ys  = bits[s] ? y1 : y0;
      red[s][0] = 0.5f * (ys - lse);                    // Re log psi
      red[s][1] = PI_F * ys * FRCP(1.0f + fabsf(ys));   // Im log psi
    }
    __syncthreads();  // S5

    if (tid == 0) {
      float re = 0.0f, im = 0.0f;
      #pragma unroll 8
      for (int s = 0; s < SORB; ++s) { re += red[s][0]; im += red[s][1]; }
      const float a = __expf(re);
      out[2 * t + 0] = a * __cosf(im);
      out[2 * t + 1] = a * __sinf(im);
    }
  }
}

// ---------------------------------------------------------------------------
extern "C" void kernel_launch(void* const* d_in, const int* in_sizes, int n_in,
                              void* d_out, int out_size, void* d_ws, size_t ws_size,
                              hipStream_t stream) {
  (void)in_sizes; (void)n_in; (void)out_size; (void)ws_size;
  const int*   x    = (const int*)  d_in[0];
  const float* W_ih = (const float*)d_in[1];
  const float* W_hh = (const float*)d_in[2];
  const float* b_ih = (const float*)d_in[3];
  const float* b_hh = (const float*)d_in[4];
  const float* fc_W = (const float*)d_in[5];
  const float* fc_b = (const float*)d_in[6];
  float* out = (float*)d_out;

  __bf16* whh_bf = (__bf16*)d_ws;  // 768*256*2 = 393216 bytes of scratch
  const int n = H3 * HDIM;
  cvt_whh_bf16<<<(n + 255) / 256, 256, 0, stream>>>(W_hh, whh_bf, n);
  gru_wavefn<<<1, NTHREADS, 0, stream>>>(x, W_ih, b_ih, b_hh, fc_W, fc_b, whh_bf, out);
}
